// STM_18786186953178
// MI455X (gfx1250) — matrix operand order repaired and checked by reading, not compiled, not run on hardware
//
#include <hip/hip_runtime.h>

// ---------------------------------------------------------------------------
// STM attention for MI455X (gfx1250): all matmuls via v_wmma_f32_16x16x32_f16.
// ---------------------------------------------------------------------------

typedef __attribute__((ext_vector_type(16))) _Float16 v16h;
typedef __attribute__((ext_vector_type(8)))  float    v8f;

union V16 { v16h v; _Float16 h[16]; uint4 u4[2]; };

constexpr int kB   = 2;
constexpr int kT   = 4;
constexpr int kC   = 256;
constexpr int kCM  = 304;
constexpr int kCK  = 32;
constexpr int kCV  = 128;
constexpr int kHW  = 4096;   // 64*64
constexpr int kTHW = 16384;  // T*HW

__device__ __forceinline__ v8f wmma16(v16h a, v16h b, v8f c) {
  return __builtin_amdgcn_wmma_f32_16x16x32_f16(false, a, false, b, (short)0, c,
                                                false, false);
}

// Load 16 halves as two contiguous 8-half (16B) runs -> v16h fragment.
__device__ __forceinline__ v16h load16h(const _Float16* p0, const _Float16* p1) {
  V16 r;
  r.u4[0] = *(const uint4*)p0;
  r.u4[1] = *(const uint4*)p1;
  return r.v;
}

// A-fragment (16x32 f16) from a row-major f32 matrix W[M][Kd].
// Lane holds row m; halves h -> k0+16*kg+h.
__device__ __forceinline__ v16h load_a_fast(const float* W, int Kd, int row,
                                            int k0, int kg) {
  const float* base = W + (size_t)row * Kd + k0 + kg * 16;
  float4 x0 = *(const float4*)(base);
  float4 x1 = *(const float4*)(base + 4);
  float4 y0 = *(const float4*)(base + 8);
  float4 y1 = *(const float4*)(base + 12);
  V16 a;
  a.h[0]=(_Float16)x0.x; a.h[1]=(_Float16)x0.y; a.h[2]=(_Float16)x0.z; a.h[3]=(_Float16)x0.w;
  a.h[4]=(_Float16)x1.x; a.h[5]=(_Float16)x1.y; a.h[6]=(_Float16)x1.z; a.h[7]=(_Float16)x1.w;
  a.h[8]=(_Float16)y0.x; a.h[9]=(_Float16)y0.y; a.h[10]=(_Float16)y0.z; a.h[11]=(_Float16)y0.w;
  a.h[12]=(_Float16)y1.x; a.h[13]=(_Float16)y1.y; a.h[14]=(_Float16)y1.z; a.h[15]=(_Float16)y1.w;
  return a.v;
}

__device__ __forceinline__ v16h load_a_guard(const float* W, int Kd, int row,
                                             int k0, int kg) {
  const float* base = W + (size_t)row * Kd;
  int ka = k0 + kg * 16, kb = k0 + kg * 16 + 8;
  V16 a;
#pragma unroll
  for (int j = 0; j < 8; ++j) {
    a.h[j]     = (ka + j < Kd) ? (_Float16)base[ka + j] : (_Float16)0.f;
    a.h[8 + j] = (kb + j < Kd) ? (_Float16)base[kb + j] : (_Float16)0.f;
  }
  return a.v;
}

// B-fragment (32x16 f16) from channel-major f32 X[Kd][N]: lane holds col p,
// halves k = 16*kg + h.
__device__ __forceinline__ v16h load_b_x(const float* X, int Kd, int N,
                                         int k0, int kg, int p) {
  V16 b;
#pragma unroll
  for (int h = 0; h < 16; ++h) {
    int c = k0 + kg * 16 + h;
    b.h[h] = (c < Kd) ? (_Float16)X[(size_t)c * N + p] : (_Float16)0.f;
  }
  return b.v;
}

// ---------------------------------------------------------------------------
// Projection (1x1 conv) GEMM: Out[m][p] = sum_c W[m][c] * X[c][p]
// MODE 0: f32 channel-major   (ldOut row stride, +t*N column offset)
// MODE 1: f16 channel-major
// MODE 2: f16 pixel-major     (Out[p][m], ld = M)
// block = 128 threads = 4 waves, each wave owns one 16-pixel tile.
// ---------------------------------------------------------------------------
template <int MODE>
__global__ __launch_bounds__(128) void proj_kernel(
    const float* __restrict__ Wm, const float* __restrict__ X,
    void* __restrict__ Out, int M, int Kd, int N, int tcount, int ldOut) {
  int lane = threadIdx.x & 31, wv = threadIdx.x >> 5;
  int n = lane & 15, kg = lane >> 4;
  int p0 = (blockIdx.x * 4 + wv) * 16;
  int m0 = blockIdx.y * 16;
  int z  = blockIdx.z;
  int b = z / tcount, t = z - b * tcount;
  const float* Xz = X + (size_t)z * Kd * N;
  int p = p0 + n;
  int rowA = m0 + n;

  v8f acc = {};
  for (int k0 = 0; k0 < Kd; k0 += 32) {
    v16h a = (k0 + 32 <= Kd) ? load_a_fast(Wm, Kd, rowA, k0, kg)
                             : load_a_guard(Wm, Kd, rowA, k0, kg);
    v16h bf = load_b_x(Xz, Kd, N, k0, kg, p);
    acc = wmma16(a, bf, acc);
  }

#pragma unroll
  for (int r = 0; r < 8; ++r) {
    int m = m0 + 8 * kg + r;
    if (MODE == 0) {
      ((float*)Out)[(size_t)b * M * ldOut + (size_t)t * N + (size_t)m * ldOut + p] = acc[r];
    } else if (MODE == 1) {
      ((_Float16*)Out)[(size_t)b * M * ldOut + (size_t)t * N + (size_t)m * ldOut + p] =
          (_Float16)acc[r];
    } else {
      ((_Float16*)Out)[(size_t)z * N * M + (size_t)p * M + m] = (_Float16)acc[r];
    }
  }
}

// ---------------------------------------------------------------------------
// Per-position l2-norm over 32 channels; write transposed (pos-major) f16.
// in: (B, 32, L) f32 channel-major; out: (B*L, 32) f16.
// ---------------------------------------------------------------------------
__global__ __launch_bounds__(256) void norm32_kernel(const float* __restrict__ in,
                                                     _Float16* __restrict__ out,
                                                     int L, int total) {
  int g = blockIdx.x * blockDim.x + threadIdx.x;
  if (g >= total) return;
  int b = g / L, pos = g - b * L;
  const float* p = in + (size_t)b * 32 * L + pos;
  float v[32];
  float s = 0.f;
#pragma unroll
  for (int c = 0; c < 32; ++c) {
    v[c] = p[(size_t)c * L];
    s += v[c] * v[c];
  }
  float inv = 1.f / fmaxf(sqrtf(s), 1e-12f);
  _Float16* o = out + (size_t)g * 32;
#pragma unroll
  for (int c = 0; c < 32; ++c) o[c] = (_Float16)(v[c] * inv);
}

// ---------------------------------------------------------------------------
// Top-4 tracking: branchless sort-network insert (4 v_cmp + 14 v_cndmask,
// no EXEC writes, no branches -> co-executes with WMMA).
// ---------------------------------------------------------------------------
struct Top4 { float v0, v1, v2, v3; int i0, i1, i2, i3; };

__device__ __forceinline__ void top4_ins(Top4& t, float v, int i) {
  bool c0 = v > t.v0, c1 = v > t.v1, c2 = v > t.v2, c3 = v > t.v3;
  float nv0 = c0 ? v    : t.v0;
  int   ni0 = c0 ? i    : t.i0;
  float nv1 = c0 ? t.v0 : (c1 ? v : t.v1);
  int   ni1 = c0 ? t.i0 : (c1 ? i : t.i1);
  float nv2 = c1 ? t.v1 : (c2 ? v : t.v2);
  int   ni2 = c1 ? t.i1 : (c2 ? i : t.i2);
  float nv3 = c2 ? t.v2 : (c3 ? v : t.v3);
  int   ni3 = c2 ? t.i2 : (c3 ? i : t.i3);
  t.v0 = nv0; t.v1 = nv1; t.v2 = nv2; t.v3 = nv3;
  t.i0 = ni0; t.i1 = ni1; t.i2 = ni2; t.i3 = ni3;
}

// ---------------------------------------------------------------------------
// Fused attention: logits (WMMA, K=CK=32) -> exp (bounded cosine, no running
// max needed) -> P via LDS -> PV accumulation (8 WMMAs over CV=128), streaming
// top-4 of logits. One wave per 16-query tile; 512 key-steps of 32.
// ---------------------------------------------------------------------------
__global__ __launch_bounds__(128) void attn_kernel(
    const _Float16* __restrict__ qk_t,   // (B, HW, 32)
    const _Float16* __restrict__ mk_t,   // (B, THW, 32)
    const _Float16* __restrict__ mv,     // (B, CV, THW)
    _Float16* __restrict__ mvo_t,        // (B, HW, CV)
    int* __restrict__ topk_out) {        // (B, HW, 4, 3)
  __shared__ __align__(16) _Float16 Pl[4][16 * 40];  // padded rows: no bank conflicts
  int lane = threadIdx.x & 31, wv = threadIdx.x >> 5;
  int n = lane & 15, kg = lane >> 4;
  int b = blockIdx.y;
  int q0 = (blockIdx.x * 4 + wv) * 16;
  _Float16* P = Pl[wv];

  const _Float16* qrow = qk_t + ((size_t)b * kHW + q0 + n) * 32;
  v16h a_qk = load16h(qrow + kg * 16, qrow + kg * 16 + 8);

  const _Float16* mkb = mk_t + (size_t)b * kTHW * 32;
  const _Float16* mvb = mv + (size_t)b * kCV * kTHW;

  v8f O[8];
  float l[8];
  Top4 tk[8];
#pragma unroll
  for (int r = 0; r < 8; ++r) {
    O[r] = (v8f){};
    l[r] = 0.f;
    tk[r].v0 = tk[r].v1 = tk[r].v2 = tk[r].v3 = -3.402823466e38f;
    tk[r].i0 = tk[r].i1 = tk[r].i2 = tk[r].i3 = 0;
  }

  for (int k0 = 0; k0 < kTHW; k0 += 32) {
    // logits for 32 keys: two 16x16 tiles
    const _Float16* r0 = mkb + (size_t)(k0 + n) * 32 + kg * 16;
    const _Float16* r1 = mkb + (size_t)(k0 + 16 + n) * 32 + kg * 16;
    v16h b0 = load16h(r0, r0 + 8);
    v16h b1 = load16h(r1, r1 + 8);
    v8f S0 = wmma16(a_qk, b0, (v8f){});
    v8f S1 = wmma16(a_qk, b1, (v8f){});

#pragma unroll
    for (int r = 0; r < 8; ++r) {
      float s0 = S0[r], s1 = S1[r];
      float e0 = __expf(s0), e1 = __expf(s1);   // s in [-1,1]: no max-shift needed
      l[r] += e0 + e1;
      top4_ins(tk[r], s0, k0 + n);
      top4_ins(tk[r], s1, k0 + 16 + n);
      P[(8 * kg + r) * 40 + n]      = (_Float16)e0;
      P[(8 * kg + r) * 40 + 16 + n] = (_Float16)e1;
    }
    asm volatile("s_wait_dscnt 0" ::: "memory");  // LDS writes visible wave-wide

    const _Float16* pr = P + n * 40;  // A-fragment row = lane&15
    v16h ap = load16h(pr + kg * 16, pr + kg * 16 + 8);
#pragma unroll
    for (int cvt = 0; cvt < 8; ++cvt) {
      const _Float16* mr = mvb + (size_t)(cvt * 16 + n) * kTHW + k0 + kg * 16;
      v16h bv = load16h(mr, mr + 8);
      O[cvt] = wmma16(ap, bv, O[cvt]);
    }
  }

  // softmax denominator: reduce partial sums across the 16 column lanes
#pragma unroll
  for (int r = 0; r < 8; ++r) {
    float s = l[r];
    s += __shfl_xor(s, 1); s += __shfl_xor(s, 2);
    s += __shfl_xor(s, 4); s += __shfl_xor(s, 8);
    l[r] = 1.f / s;
  }

  // write normalized O as f16, pixel-major (B, HW, CV)
#pragma unroll
  for (int r = 0; r < 8; ++r) {
    int q = q0 + 8 * kg + r;
    _Float16* orow = mvo_t + ((size_t)b * kHW + q) * kCV + n;
#pragma unroll
    for (int cvt = 0; cvt < 8; ++cvt)
      orow[cvt * 16] = (_Float16)(O[cvt][r] * l[r]);
  }

  // merge per-lane top-4 across the 16 column lanes (branchless)
#pragma unroll
  for (int mask = 1; mask < 16; mask <<= 1) {
#pragma unroll
    for (int r = 0; r < 8; ++r) {
      float ov0 = __shfl_xor(tk[r].v0, mask), ov1 = __shfl_xor(tk[r].v1, mask);
      float ov2 = __shfl_xor(tk[r].v2, mask), ov3 = __shfl_xor(tk[r].v3, mask);
      int oi0 = __shfl_xor(tk[r].i0, mask), oi1 = __shfl_xor(tk[r].i1, mask);
      int oi2 = __shfl_xor(tk[r].i2, mask), oi3 = __shfl_xor(tk[r].i3, mask);
      top4_ins(tk[r], ov0, oi0); top4_ins(tk[r], ov1, oi1);
      top4_ins(tk[r], ov2, oi2); top4_ins(tk[r], ov3, oi3);
    }
  }
  if (n == 0) {
#pragma unroll
    for (int r = 0; r < 8; ++r) {
      int q = q0 + 8 * kg + r;
      int* o = topk_out + ((size_t)b * kHW + q) * 4 * 3;
      int id[4] = {tk[r].i0, tk[r].i1, tk[r].i2, tk[r].i3};
#pragma unroll
      for (int j = 0; j < 4; ++j) {
        int k = id[j];
        int t = k >> 12;        // / HW
        int rem = k & 4095;
        o[j * 3 + 0] = t;
        o[j * 3 + 1] = rem >> 6;  // / W
        o[j * 3 + 2] = rem & 63;  // % W
      }
    }
  }
}

// ---------------------------------------------------------------------------
// Epilogue: y = relu(BN(W_smooth @ cat(qv, mv_out))) with cat in pixel-major
// f16; 256x256 GEMM in 8 WMMA steps per tile.
// ---------------------------------------------------------------------------
__global__ __launch_bounds__(128) void smooth_bn_kernel(
    const float* __restrict__ Ws,       // (256, 256)
    const _Float16* __restrict__ qv_t,  // (B, HW, 128)
    const _Float16* __restrict__ mvo_t, // (B, HW, 128)
    const float* __restrict__ gma, const float* __restrict__ bta,
    const float* __restrict__ mu, const float* __restrict__ var,
    float* __restrict__ y) {            // (B, 256, HW)
  int lane = threadIdx.x & 31, wv = threadIdx.x >> 5;
  int n = lane & 15, kg = lane >> 4;
  int b = blockIdx.z;
  int m0 = blockIdx.y * 16;
  int p0 = (blockIdx.x * 4 + wv) * 16;
  int p = p0 + n;
  int rowA = m0 + n;

  const _Float16* qrow = qv_t + ((size_t)b * kHW + p) * kCV;
  const _Float16* mrow = mvo_t + ((size_t)b * kHW + p) * kCV;

  v8f acc = {};
#pragma unroll
  for (int k0 = 0; k0 < 256; k0 += 32) {
    v16h a = load_a_fast(Ws, 256, rowA, k0, kg);
    const _Float16* src = (k0 < 128) ? (qrow + k0 + kg * 16)
                                     : (mrow + (k0 - 128) + kg * 16);
    v16h bf = load16h(src, src + 8);
    acc = wmma16(a, bf, acc);
  }

#pragma unroll
  for (int r = 0; r < 8; ++r) {
    int o = m0 + 8 * kg + r;
    float sc = gma[o] * rsqrtf(var[o] + 1e-5f);
    float v = (acc[r] - mu[o]) * sc + bta[o];
    y[((size_t)b * kC + o) * kHW + p] = fmaxf(v, 0.f);
  }
}

// ---------------------------------------------------------------------------
extern "C" void kernel_launch(void* const* d_in, const int* in_sizes, int n_in,
                              void* d_out, int out_size, void* d_ws, size_t ws_size,
                              hipStream_t stream) {
  const float* x        = (const float*)d_in[0];
  const float* mem      = (const float*)d_in[1];
  const float* w_qk     = (const float*)d_in[2];
  const float* w_qv     = (const float*)d_in[3];
  const float* w_mk     = (const float*)d_in[4];
  const float* w_mv     = (const float*)d_in[5];
  const float* w_smooth = (const float*)d_in[6];
  const float* bn_gamma = (const float*)d_in[7];
  const float* bn_beta  = (const float*)d_in[8];
  const float* bn_mean  = (const float*)d_in[9];
  const float* bn_var   = (const float*)d_in[10];

  float* y_out = (float*)d_out;
  int* ids_out = (int*)(y_out + (size_t)kB * kC * kHW);

  // workspace carve (aligned to 256B)
  char* wsp = (char*)d_ws;
  auto carve = [&](size_t bytes) {
    char* pp = wsp;
    wsp += (bytes + 255) & ~(size_t)255;
    return pp;
  };
  float*    qk_f32 = (float*)carve((size_t)kB * kCK * kHW * 4);
  float*    mk_f32 = (float*)carve((size_t)kB * kCK * kTHW * 4);
  _Float16* qk_t   = (_Float16*)carve((size_t)kB * kHW * kCK * 2);
  _Float16* mk_t   = (_Float16*)carve((size_t)kB * kTHW * kCK * 2);
  _Float16* qv_t   = (_Float16*)carve((size_t)kB * kHW * kCV * 2);
  _Float16* mv_cm  = (_Float16*)carve((size_t)kB * kCV * kTHW * 2);
  _Float16* mvo_t  = (_Float16*)carve((size_t)kB * kHW * kCV * 2);

  dim3 blk(128);
  // projections
  proj_kernel<0><<<dim3(64, 2, kB), blk, 0, stream>>>(w_qk, x, qk_f32, kCK, kC, kHW, 1, kHW);
  proj_kernel<2><<<dim3(64, 8, kB), blk, 0, stream>>>(w_qv, x, qv_t, kCV, kC, kHW, 1, kCV);
  proj_kernel<0><<<dim3(64, 2, kB * kT), blk, 0, stream>>>(w_mk, mem, mk_f32, kCK, kCM, kHW, kT, kTHW);
  proj_kernel<1><<<dim3(64, 8, kB * kT), blk, 0, stream>>>(w_mv, mem, mv_cm, kCV, kCM, kHW, kT, kTHW);
  // l2-norm + transpose to f16
  {
    int tot1 = kB * kHW;
    norm32_kernel<<<dim3((tot1 + 255) / 256), dim3(256), 0, stream>>>(qk_f32, qk_t, kHW, tot1);
    int tot2 = kB * kTHW;
    norm32_kernel<<<dim3((tot2 + 255) / 256), dim3(256), 0, stream>>>(mk_f32, mk_t, kTHW, tot2);
  }
  // fused attention + top-4
  attn_kernel<<<dim3(64, kB), blk, 0, stream>>>(qk_t, mk_t, mv_cm, mvo_t, ids_out);
  // smooth conv + BN + ReLU
  smooth_bn_kernel<<<dim3(64, 16, kB), blk, 0, stream>>>(w_smooth, qv_t, mvo_t,
                                                         bn_gamma, bn_beta, bn_mean,
                                                         bn_var, y_out);
}
